// GCN_11982958756665
// MI455X (gfx1250) — compile-verified, matrix operand order checked by reading, not run on hardware
//
#include <hip/hip_runtime.h>
#include <hip/hip_bf16.h>
#include <stdint.h>

#define NNODES 50000
#define NEDGES 800000
#define DFEAT  64
#define DHID   16

typedef __attribute__((ext_vector_type(16))) _Float16 v16h;
typedef __attribute__((ext_vector_type(8)))  float    v8f;

// ---------------- degree / normalization ----------------

__global__ void k_deg_init(float* __restrict__ deg) {
    int i = blockIdx.x * blockDim.x + threadIdx.x;
    if (i < NNODES) deg[i] = 1.0f;               // self-loop contributes 1
}

__global__ void k_deg_edges(const int64_t* __restrict__ dst, float* __restrict__ deg) {
    int e = blockIdx.x * blockDim.x + threadIdx.x;
    if (e < NEDGES) atomicAdd(&deg[(int)dst[e]], 1.0f);
}

__global__ void k_dinv(float* __restrict__ deg) {
    int i = blockIdx.x * blockDim.x + threadIdx.x;
    if (i < NNODES) deg[i] = rsqrtf(deg[i]);     // deg >= 1 always (self-loop)
}

// ---------------- WMMA GEMM 1: H1[N,16] = x[N,64] @ W1[64,16] ----------------
// One wave per 16-row tile. A: lane&15 = row, (lane>>4) selects K-half.
// 16-bit A layout (ISA 7.12.2): VGPR v, half h -> K = (v>=4?16:0) + 8*kgrp + 2*(v&3) + h.
// B uses the mirrored layout with lane&15 = column.

__global__ void __launch_bounds__(256) k_gemm1(const float* __restrict__ x,
                                               const float* __restrict__ W1,
                                               float* __restrict__ H1) {
    const int wid  = (blockIdx.x * blockDim.x + threadIdx.x) >> 5;
    const int lane = threadIdx.x & 31;
    if (wid >= NNODES / 16) return;              // wave-uniform: EXEC stays all-1s
    const int rc   = lane & 15;                  // row for A / col for B,D
    const int kgrp = (lane >> 4) & 1;
    const float* xr = x + (wid * 16 + rc) * DFEAT;

    v8f c = {};
#pragma unroll
    for (int ks = 0; ks < 2; ++ks) {             // K = 64 -> two 16x16x32 steps
        const int k0 = ks * 32;
        v16h a, b;
#pragma unroll
        for (int v = 0; v < 8; ++v) {
            const int kb = k0 + ((v >> 2) << 4) + (kgrp << 3) + ((v & 3) << 1);
            a[2 * v + 0] = (_Float16)xr[kb + 0];
            a[2 * v + 1] = (_Float16)xr[kb + 1];
            b[2 * v + 0] = (_Float16)W1[(kb + 0) * DHID + rc];
            b[2 * v + 1] = (_Float16)W1[(kb + 1) * DHID + rc];
        }
        c = __builtin_amdgcn_wmma_f32_16x16x32_f16(false, a, false, b,
                                                   (short)0, c, false, false);
    }
    // D layout: VGPR r -> row M = r + 8*kgrp, col = lane&15
    float* outp = H1 + (wid * 16 + (kgrp << 3)) * DHID + rc;
#pragma unroll
    for (int r = 0; r < 8; ++r) outp[r * DHID] = c[r];
}

// ---------------- WMMA GEMM 2: H2[N,64] = Z[N,16] @ W2[16,64] ----------------
// K = 16 zero-padded to 32; one wave per 16x16 output tile (4 col tiles/row tile).

__global__ void __launch_bounds__(256) k_gemm2(const float* __restrict__ Z,
                                               const float* __restrict__ W2,
                                               float* __restrict__ H2) {
    const int wid  = (blockIdx.x * blockDim.x + threadIdx.x) >> 5;
    const int lane = threadIdx.x & 31;
    if (wid >= (NNODES / 16) * (DFEAT / 16)) return;
    const int tileM = wid >> 2;
    const int tileN = wid & 3;
    const int rc    = lane & 15;
    const int kgrp  = (lane >> 4) & 1;
    const float* zr = Z + (tileM * 16 + rc) * DHID;

    v16h a, b;
#pragma unroll
    for (int v = 0; v < 8; ++v) {
        const int kb = ((v >> 2) << 4) + (kgrp << 3) + ((v & 3) << 1);
        const bool in0 = (kb + 0) < DHID, in1 = (kb + 1) < DHID;
        a[2 * v + 0] = in0 ? (_Float16)zr[kb + 0] : (_Float16)0.0f;
        a[2 * v + 1] = in1 ? (_Float16)zr[kb + 1] : (_Float16)0.0f;
        b[2 * v + 0] = in0 ? (_Float16)W2[(kb + 0) * DFEAT + tileN * 16 + rc] : (_Float16)0.0f;
        b[2 * v + 1] = in1 ? (_Float16)W2[(kb + 1) * DFEAT + tileN * 16 + rc] : (_Float16)0.0f;
    }
    v8f c = {};
    c = __builtin_amdgcn_wmma_f32_16x16x32_f16(false, a, false, b,
                                               (short)0, c, false, false);
    float* outp = H2 + (tileM * 16 + (kgrp << 3)) * DFEAT + tileN * 16 + rc;
#pragma unroll
    for (int r = 0; r < 8; ++r) outp[r * DFEAT] = c[r];
}

// ---------------- aggregation (self-loop + bias init, then edge atomics) ----------------

__global__ void k_agg1_init(const float* __restrict__ H1, const float* __restrict__ dinv,
                            const float* __restrict__ b1, float* __restrict__ A1) {
    int i = blockIdx.x * blockDim.x + threadIdx.x;
    if (i < NNODES * DHID) {
        int node = i / DHID, f = i - node * DHID;
        float di = dinv[node];
        A1[i] = H1[i] * di * di + b1[f];         // self-loop norm = dinv^2; bias folded in
    }
}

__global__ void k_agg1_edges(const int64_t* __restrict__ src, const int64_t* __restrict__ dst,
                             const float* __restrict__ dinv, const float* __restrict__ H1,
                             float* __restrict__ A1) {
    int e = blockIdx.x * blockDim.x + threadIdx.x;
    if (e >= NEDGES) return;
    int s = (int)src[e], d = (int)dst[e];
    float nrm = dinv[s] * dinv[d];
    const float4* hp = (const float4*)(H1 + s * DHID);
    float* op = A1 + d * DHID;
#pragma unroll
    for (int q = 0; q < 4; ++q) {
        float4 h = hp[q];
        atomicAdd(op + q * 4 + 0, h.x * nrm);
        atomicAdd(op + q * 4 + 1, h.y * nrm);
        atomicAdd(op + q * 4 + 2, h.z * nrm);
        atomicAdd(op + q * 4 + 3, h.w * nrm);
    }
}

__global__ void k_relu(float* __restrict__ A1) {
    int i = blockIdx.x * blockDim.x + threadIdx.x;
    if (i < NNODES * DHID) A1[i] = fmaxf(A1[i], 0.0f);
}

__global__ void k_agg2_init(const float* __restrict__ H2, const float* __restrict__ dinv,
                            const float* __restrict__ b2, float* __restrict__ out) {
    int i = blockIdx.x * blockDim.x + threadIdx.x;
    if (i < NNODES * DFEAT) {
        int node = i / DFEAT, f = i - node * DFEAT;
        float di = dinv[node];
        out[i] = H2[i] * di * di + b2[f];
    }
}

__global__ void k_agg2_edges(const int64_t* __restrict__ src, const int64_t* __restrict__ dst,
                             const float* __restrict__ dinv, const float* __restrict__ H2,
                             float* __restrict__ out) {
    int t = blockIdx.x * blockDim.x + threadIdx.x;
    if (t >= NEDGES * 4) return;
    int e = t >> 2, chunk = t & 3;               // 4 threads/edge, 16 feats each
    int s = (int)src[e], d = (int)dst[e];
    float nrm = dinv[s] * dinv[d];
    const float4* hp = (const float4*)(H2 + s * DFEAT + chunk * 16);
    float* op = out + d * DFEAT + chunk * 16;
#pragma unroll
    for (int q = 0; q < 4; ++q) {
        float4 h = hp[q];
        atomicAdd(op + q * 4 + 0, h.x * nrm);
        atomicAdd(op + q * 4 + 1, h.y * nrm);
        atomicAdd(op + q * 4 + 2, h.z * nrm);
        atomicAdd(op + q * 4 + 3, h.w * nrm);
    }
}

// ---------------- launch ----------------

extern "C" void kernel_launch(void* const* d_in, const int* in_sizes, int n_in,
                              void* d_out, int out_size, void* d_ws, size_t ws_size,
                              hipStream_t stream) {
    const float*   x   = (const float*)d_in[0];
    const int64_t* ei  = (const int64_t*)d_in[1];     // [2, E] int64
    const float*   W1  = (const float*)d_in[2];
    const float*   b1  = (const float*)d_in[3];
    const float*   W2  = (const float*)d_in[4];
    const float*   b2  = (const float*)d_in[5];
    float*         out = (float*)d_out;

    const int64_t* src = ei;                 // edge_index[0]
    const int64_t* dst = ei + NEDGES;        // edge_index[1]

    // workspace layout (floats): dinv | H1 | AGG1/Z | H2  == 50000*97*4 B ~ 19.4 MB
    float* dinv = (float*)d_ws;
    float* H1   = dinv + NNODES;
    float* A1   = H1 + (size_t)NNODES * DHID;
    float* H2   = A1 + (size_t)NNODES * DHID;

    const int T = 256;
    k_deg_init <<<(NNODES + T - 1) / T, T, 0, stream>>>(dinv);
    k_deg_edges<<<(NEDGES + T - 1) / T, T, 0, stream>>>(dst, dinv);
    k_dinv     <<<(NNODES + T - 1) / T, T, 0, stream>>>(dinv);

    k_gemm1<<<((NNODES / 16) + 7) / 8, T, 0, stream>>>(x, W1, H1);

    k_agg1_init <<<(NNODES * DHID + T - 1) / T, T, 0, stream>>>(H1, dinv, b1, A1);
    k_agg1_edges<<<(NEDGES + T - 1) / T, T, 0, stream>>>(src, dst, dinv, H1, A1);
    k_relu      <<<(NNODES * DHID + T - 1) / T, T, 0, stream>>>(A1);

    k_gemm2<<<((NNODES / 16) * (DFEAT / 16) + 7) / 8, T, 0, stream>>>(A1, W2, H2);

    k_agg2_init <<<(NNODES * DFEAT + T - 1) / T, T, 0, stream>>>(H2, dinv, b2, out);
    k_agg2_edges<<<(NEDGES * 4 + T - 1) / T, T, 0, stream>>>(src, dst, dinv, H2, out);
}
